// AttnDecoderRNN_54065048322492
// MI455X (gfx1250) — compile-verified, matrix operand order checked by reading, not run on hardware
//
#include <hip/hip_runtime.h>
#include <math.h>

// ---------------------------------------------------------------------------
// AttnDecoderRNN for MI455X (gfx1250, wave32, WMMA).
// B=64, S=128, T=32, H=1024, V=32000.
// GEMMs via v_wmma_f32_16x16x32_bf16 (bf16 inputs, f32 accumulate).
// Weights converted to bf16 once per launch; W_out(bf16)=65.5MB stays in the
// 192MB L2 across the 32 sequential steps.
//
// GEMM tile shape: each wave owns 64(M) x 32(N): one B fragment feeds 4 WMMA
// against 4 resident A fragments -> W (the streamed operand) is read exactly
// once per step, and each k-step is [12 b128-load clause -> 8 WMMA chain].
// ---------------------------------------------------------------------------

typedef __attribute__((ext_vector_type(16))) __bf16 v16bf;
typedef __attribute__((ext_vector_type(8)))  __bf16 v8bf;
typedef __attribute__((ext_vector_type(8)))  float  v8f;

union BF16x16 { v16bf v; v8bf h[2]; };

#define SOS_TOKEN 1

// ---------------------------------------------------------------------------
// Generic GEMM: C[M,N] = A[M,K] * W[N,K]^T + bias[N]
// A, W bf16 row-major; C f32 row-major.
// Block = 128 threads = 4 waves; each wave: 64(M) x 32(N) tile.
// grid = dim3(N/128, M/64). Requires K%32==0, N%128==0, M%64==0.
// WMMA fragment layout per CDNA5 ISA 7.12.2 (16-bit A 16x32: lane m=l16,
// elements [0..7] at K = k+8*hf, [8..15] at K = k+16+8*hf; B mirrors the K
// striping with lane<->N since B = W^T; C/D f32: VGPR r <-> M=r+8*hf).
// ---------------------------------------------------------------------------
__global__ __launch_bounds__(128)
void ad_gemm_bf16_xwt(const __bf16* __restrict__ A,
                      const __bf16* __restrict__ W,
                      const float* __restrict__ bias,
                      float* __restrict__ C,
                      int M, int N, int K) {
  const int lane = threadIdx.x & 31;
  const int wave = threadIdx.x >> 5;
  const int hf   = lane >> 4;    // half-wave: 0 or 1
  const int l16  = lane & 15;
  const int m0 = blockIdx.y * 64;
  const int n0 = blockIdx.x * 128 + wave * 32;   // two 16-wide N strips

  const v8f vz = {0.f, 0.f, 0.f, 0.f, 0.f, 0.f, 0.f, 0.f};
  v8f acc[2][4];   // [n-strip][m-tile]
#pragma unroll
  for (int j = 0; j < 2; ++j)
#pragma unroll
    for (int i = 0; i < 4; ++i) acc[j][i] = vz;

  const __bf16* a0 = A + (size_t)(m0 + l16) * K + hf * 8;  // m-tile 0 base
  const __bf16* w0 = W + (size_t)(n0 + l16) * K + hf * 8;  // n-strip 0 base
  const __bf16* w1 = w0 + (size_t)16 * K;                  // n-strip 1 base

#pragma unroll 2
  for (int k = 0; k < K; k += 32) {
    BF16x16 a[4], b[2];
#pragma unroll
    for (int i = 0; i < 4; ++i) {
      const __bf16* p = a0 + (size_t)(16 * i) * K + k;
      a[i].h[0] = *(const v8bf*)(p);
      a[i].h[1] = *(const v8bf*)(p + 16);
    }
    b[0].h[0] = *(const v8bf*)(w0 + k);
    b[0].h[1] = *(const v8bf*)(w0 + k + 16);
    b[1].h[0] = *(const v8bf*)(w1 + k);
    b[1].h[1] = *(const v8bf*)(w1 + k + 16);

    // keep the W stream warm ~16 k-iterations ahead (global_prefetch_b8)
    __builtin_prefetch(w0 + k + 512, 0, 1);
    __builtin_prefetch(w1 + k + 512, 0, 1);

#pragma unroll
    for (int j = 0; j < 2; ++j)
#pragma unroll
      for (int i = 0; i < 4; ++i)
        acc[j][i] = __builtin_amdgcn_wmma_f32_16x16x32_bf16(
            /*neg_a=*/false, a[i].v, /*neg_b=*/false, b[j].v,
            /*c_mod=*/(short)0, acc[j][i],
            /*reuse_a=*/false, /*reuse_b=*/false);
  }

#pragma unroll
  for (int j = 0; j < 2; ++j) {
    const int n = n0 + j * 16 + l16;
    const float bv = bias[n];
#pragma unroll
    for (int i = 0; i < 4; ++i) {
#pragma unroll
      for (int r = 0; r < 8; ++r) {
        const int m = m0 + i * 16 + r + 8 * hf;
        C[(size_t)m * N + n] = acc[j][i][r] + bv;
      }
    }
  }
}

// ---------------------------------------------------------------------------
// f32 -> bf16 conversion
// ---------------------------------------------------------------------------
__global__ void ad_cvt_bf16(const float* __restrict__ src,
                            __bf16* __restrict__ dst, size_t n) {
  size_t i = (size_t)blockIdx.x * blockDim.x + threadIdx.x;
  if (i < n) dst[i] = (__bf16)src[i];
}

// h0 = encoder_hidden; also produce bf16 copy
__global__ void ad_init_h(const float* __restrict__ eh,
                          float* __restrict__ h,
                          __bf16* __restrict__ hbf, int n) {
  int i = blockIdx.x * blockDim.x + threadIdx.x;
  if (i < n) { float v = eh[i]; h[i] = v; hbf[i] = (__bf16)v; }
}

// x_bf[:, 0:H] = bf16(embedding[token])
__global__ void ad_embed(const int* __restrict__ target,
                         const float* __restrict__ embedding,
                         __bf16* __restrict__ x_bf,
                         int t, int T, int H) {
  int idx = blockIdx.x * blockDim.x + threadIdx.x;   // B*H threads
  int b = idx / H, h = idx - b * H;
  int tok = (t == 0) ? SOS_TOKEN : target[b * T + (t - 1)];
  x_bf[(size_t)b * 2 * H + h] = (__bf16)embedding[(size_t)tok * H + h];
}

// scores[b,s] = dot(Va, tanh(q[b] + keys[b,s])) + bv   (one wave per row)
__global__ __launch_bounds__(128)
void ad_attn_scores(const float* __restrict__ q,
                    const float* __restrict__ keys,
                    const float* __restrict__ Va,
                    const float* __restrict__ bv,
                    float* __restrict__ scores, int S, int H) {
  int row = blockIdx.x * 4 + (threadIdx.x >> 5);   // row = b*S + s
  int lane = threadIdx.x & 31;
  int b = row / S;
  const float* qb = q + (size_t)b * H;
  const float* kb = keys + (size_t)row * H;
  float acc = 0.f;
  for (int h = lane; h < H; h += 32)
    acc += Va[h] * tanhf(qb[h] + kb[h]);
  for (int off = 16; off > 0; off >>= 1)
    acc += __shfl_xor(acc, off, 32);
  if (lane == 0) scores[row] = acc + bv[0];
}

// softmax over S=128 per batch row; writes attention weights + d_out slice
__global__ __launch_bounds__(128)
void ad_softmax(const float* __restrict__ scores,
                float* __restrict__ w,
                float* __restrict__ att_out,
                int S, int t, int T) {
  __shared__ float red[128];
  int b = blockIdx.x, s = threadIdx.x;
  float v = scores[b * S + s];
  red[s] = v; __syncthreads();
  for (int off = 64; off > 0; off >>= 1) {
    if (s < off) red[s] = fmaxf(red[s], red[s + off]);
    __syncthreads();
  }
  float mx = red[0]; __syncthreads();
  float e = __expf(v - mx);
  red[s] = e; __syncthreads();
  for (int off = 64; off > 0; off >>= 1) {
    if (s < off) red[s] += red[s + off];
    __syncthreads();
  }
  float wv = e / red[0];
  w[b * S + s] = wv;
  att_out[((size_t)b * T + t) * S + s] = wv;
}

// x_bf[:, H:2H] = bf16( sum_s w[b,s] * enc[b,s,:] )
__global__ void ad_context(const float* __restrict__ w,
                           const float* __restrict__ enc,
                           __bf16* __restrict__ x_bf, int S, int H) {
  int idx = blockIdx.x * blockDim.x + threadIdx.x;   // B*H threads
  int b = idx / H, h = idx - b * H;
  const float* wb = w + (size_t)b * S;
  const float* eb = enc + (size_t)b * S * H + h;
  float acc = 0.f;
  for (int s = 0; s < S; ++s)
    acc += wb[s] * eb[(size_t)s * H];
  x_bf[(size_t)b * 2 * H + H + h] = (__bf16)acc;
}

// GRU gates: r,z,n fusion -> h_new (f32 + bf16); optional h_final output
__global__ void ad_gru_gate(const float* __restrict__ gi,
                            const float* __restrict__ gh,
                            const float* __restrict__ h,
                            float* __restrict__ h_new,
                            __bf16* __restrict__ h_new_bf,
                            float* __restrict__ h_final,
                            int H, int write_final) {
  int idx = blockIdx.x * blockDim.x + threadIdx.x;   // B*H threads
  int b = idx / H, j = idx - b * H;
  const float* gib = gi + (size_t)b * 3 * H;
  const float* ghb = gh + (size_t)b * 3 * H;
  float r = 1.f / (1.f + __expf(-(gib[j] + ghb[j])));
  float z = 1.f / (1.f + __expf(-(gib[H + j] + ghb[H + j])));
  float n = tanhf(gib[2 * H + j] + r * ghb[2 * H + j]);
  float hn = (1.f - z) * n + z * h[idx];
  h_new[idx] = hn;
  h_new_bf[idx] = (__bf16)hn;
  if (write_final) h_final[idx] = hn;
}

// log_softmax over V per batch row -> d_out log_probs[b, t, :]
__global__ __launch_bounds__(256)
void ad_log_softmax(const float* __restrict__ logits,
                    float* __restrict__ out, int V, int t, int T) {
  __shared__ float red[256];
  int b = blockIdx.x, tid = threadIdx.x;
  const float* row = logits + (size_t)b * V;
  float mx = -INFINITY;
  for (int v = tid; v < V; v += 256) mx = fmaxf(mx, row[v]);
  red[tid] = mx; __syncthreads();
  for (int off = 128; off > 0; off >>= 1) {
    if (tid < off) red[tid] = fmaxf(red[tid], red[tid + off]);
    __syncthreads();
  }
  mx = red[0]; __syncthreads();
  float s = 0.f;
  for (int v = tid; v < V; v += 256) s += __expf(row[v] - mx);
  red[tid] = s; __syncthreads();
  for (int off = 128; off > 0; off >>= 1) {
    if (tid < off) red[tid] += red[tid + off];
    __syncthreads();
  }
  float lse = mx + logf(red[0]);
  float* o = out + ((size_t)b * T + t) * V;
  for (int v = tid; v < V; v += 256) o[v] = row[v] - lse;
}

// ---------------------------------------------------------------------------
extern "C" void kernel_launch(void* const* d_in, const int* in_sizes, int n_in,
                              void* d_out, int out_size, void* d_ws, size_t ws_size,
                              hipStream_t stream) {
  (void)in_sizes; (void)n_in; (void)out_size; (void)ws_size;
  const int B = 64, S = 128, T = 32, H = 1024, V = 32000;

  const float* enc  = (const float*)d_in[0];
  const float* ehid = (const float*)d_in[1];
  const int*   tgt  = (const int*)d_in[2];
  const float* emb  = (const float*)d_in[3];
  const float* Wa   = (const float*)d_in[4];
  const float* ba   = (const float*)d_in[5];
  const float* Ua   = (const float*)d_in[6];
  const float* bu   = (const float*)d_in[7];
  const float* Va   = (const float*)d_in[8];
  const float* bv   = (const float*)d_in[9];
  const float* Wih  = (const float*)d_in[10];
  const float* bih  = (const float*)d_in[11];
  const float* Whh  = (const float*)d_in[12];
  const float* bhh  = (const float*)d_in[13];
  const float* Wout = (const float*)d_in[14];
  const float* bout = (const float*)d_in[15];

  float* out_logp = (float*)d_out;                        // (B,T,V)
  float* out_hfin = out_logp + (size_t)B * T * V;         // (1,B,H)
  float* out_attn = out_hfin + (size_t)B * H;             // (B,T,S)

  // ---- carve workspace (all re-derived every call: deterministic) ----
  char* ws = (char*)d_ws;
  size_t off = 0;
  auto carve = [&](size_t bytes) -> void* {
    void* p = ws + off;
    off = (off + bytes + 255) & ~(size_t)255;
    return p;
  };
  __bf16* enc_bf  = (__bf16*)carve((size_t)B * S * H * 2);
  __bf16* Wa_bf   = (__bf16*)carve((size_t)H * H * 2);
  __bf16* Ua_bf   = (__bf16*)carve((size_t)H * H * 2);
  __bf16* Wih_bf  = (__bf16*)carve((size_t)3 * H * 2 * H * 2);
  __bf16* Whh_bf  = (__bf16*)carve((size_t)3 * H * H * 2);
  __bf16* Wout_bf = (__bf16*)carve((size_t)V * H * 2);
  float* keys     = (float*)carve((size_t)B * S * H * 4);
  float* qbuf     = (float*)carve((size_t)B * H * 4);
  float* scores   = (float*)carve((size_t)B * S * 4);
  float* wbuf     = (float*)carve((size_t)B * S * 4);
  __bf16* x_bf    = (__bf16*)carve((size_t)B * 2 * H * 2);
  float* gi       = (float*)carve((size_t)B * 3 * H * 4);
  float* gh       = (float*)carve((size_t)B * 3 * H * 4);
  float* hf[2]; __bf16* hb[2];
  hf[0] = (float*)carve((size_t)B * H * 4);
  hf[1] = (float*)carve((size_t)B * H * 4);
  hb[0] = (__bf16*)carve((size_t)B * H * 2);
  hb[1] = (__bf16*)carve((size_t)B * H * 2);
  float* logits   = (float*)carve((size_t)B * V * 4);

  auto cvt = [&](const float* s, __bf16* d, size_t n) {
    ad_cvt_bf16<<<(unsigned)((n + 255) / 256), 256, 0, stream>>>(s, d, n);
  };

  // ---- one-time per launch: bf16 weights, h0, keys_proj ----
  cvt(enc,  enc_bf,  (size_t)B * S * H);
  cvt(Wa,   Wa_bf,   (size_t)H * H);
  cvt(Ua,   Ua_bf,   (size_t)H * H);
  cvt(Wih,  Wih_bf,  (size_t)3 * H * 2 * H);
  cvt(Whh,  Whh_bf,  (size_t)3 * H * H);
  cvt(Wout, Wout_bf, (size_t)V * H);

  ad_init_h<<<(B * H + 255) / 256, 256, 0, stream>>>(ehid, hf[0], hb[0], B * H);

  // keys_proj = enc @ Ua^T + bu : M=B*S=8192, N=H, K=H
  ad_gemm_bf16_xwt<<<dim3(H / 128, (B * S) / 64), 128, 0, stream>>>(
      enc_bf, Ua_bf, bu, keys, B * S, H, H);

  int cur = 0;
  for (int t = 0; t < T; ++t) {
    int nxt = cur ^ 1;

    // x_bf[:, 0:H] = emb(token_t)
    ad_embed<<<(B * H) / 256, 256, 0, stream>>>(tgt, emb, x_bf, t, T, H);

    // q = h @ Wa^T + ba : M=64, N=1024, K=1024
    ad_gemm_bf16_xwt<<<dim3(H / 128, B / 64), 128, 0, stream>>>(
        hb[cur], Wa_bf, ba, qbuf, B, H, H);

    // scores = Va . tanh(q + keys) + bv
    ad_attn_scores<<<(B * S) / 4, 128, 0, stream>>>(qbuf, keys, Va, bv,
                                                    scores, S, H);

    // softmax over S -> weights + attentions output
    ad_softmax<<<B, 128, 0, stream>>>(scores, wbuf, out_attn, S, t, T);

    // x_bf[:, H:2H] = w @ enc
    ad_context<<<(B * H) / 256, 256, 0, stream>>>(wbuf, enc, x_bf, S, H);

    // gi = [emb, ctx] @ W_ih^T + b_ih : M=64, N=3072, K=2048
    ad_gemm_bf16_xwt<<<dim3((3 * H) / 128, B / 64), 128, 0, stream>>>(
        x_bf, Wih_bf, bih, gi, B, 3 * H, 2 * H);

    // gh = h @ W_hh^T + b_hh : M=64, N=3072, K=1024
    ad_gemm_bf16_xwt<<<dim3((3 * H) / 128, B / 64), 128, 0, stream>>>(
        hb[cur], Whh_bf, bhh, gh, B, 3 * H, H);

    // GRU fuse -> h_new (also h_final at last step)
    ad_gru_gate<<<(B * H) / 256, 256, 0, stream>>>(
        gi, gh, hf[cur], hf[nxt], hb[nxt], out_hfin, H, (t == T - 1) ? 1 : 0);

    // logits = h_new @ W_out^T + b_out : M=64, N=32000, K=1024
    ad_gemm_bf16_xwt<<<dim3(V / 128, B / 64), 128, 0, stream>>>(
        hb[nxt], Wout_bf, bout, logits, B, V, H);

    // log_softmax -> d_out log_probs[:, t, :]
    ad_log_softmax<<<B, 256, 0, stream>>>(logits, out_logp, V, t, T);

    cur = nxt;
  }
}
